// MultiHeadAttention_23527830847633
// MI455X (gfx1250) — matrix operand order repaired and checked by reading, not compiled, not run on hardware
//
#include <hip/hip_runtime.h>

// ---------------------------------------------------------------------------
// MI455X (gfx1250) multi-head attention with RoPE.
// bf16 WMMA pipeline + TDM (tensor_load_to_lds) staging of the GEMM A-tile.
// Layouts follow CDNA5 ISA fragment tables (wave32).
// ---------------------------------------------------------------------------

#define D_MODEL 1024
#define SEQ     2048
#define BATCH   2
#define NHEAD   16
#define DK      64
#define ROWS    (BATCH*SEQ)          // 4096 token rows

typedef __attribute__((ext_vector_type(16))) __bf16 v16bf;
typedef __attribute__((ext_vector_type(8)))  float  v8f;
typedef __attribute__((ext_vector_type(4)))  unsigned int v4u;
typedef __attribute__((ext_vector_type(8)))  int    v8i;
typedef __attribute__((ext_vector_type(4)))  int    v4i;

struct alignas(16) U128 { unsigned int w[4]; };

union F8  { v8f  v; float f[8]; };
union AB  { v16bf v; U128 q[2]; unsigned int u[8]; };
union PK8 { U128 q; unsigned short s[8]; };

__device__ __forceinline__ unsigned short f2bf(float f) {
    union { float f; unsigned u; } x; x.f = f;
    unsigned r = x.u + 0x7fffu + ((x.u >> 16) & 1u);   // round-to-nearest-even
    return (unsigned short)(r >> 16);
}

// A fragment, 16x32 bf16 (M x K), row-major source with leading dim ld.
// lane<16: row=lane, K = k0+{0..7, 16..23}; lane>=16: row=lane-16, K=+8 offset.
__device__ __forceinline__ v16bf load_A_frag(const unsigned short* base,
                                             int row0, int k0, int ld, int lane) {
    int r    = row0 + (lane & 15);
    int kofs = (lane & 16) ? 8 : 0;
    const unsigned short* p = base + (size_t)r * ld + k0 + kofs;
    AB u;
    u.q[0] = *(const U128*)(p);        // K = k0+kofs   .. +7
    u.q[1] = *(const U128*)(p + 16);   // K = k0+kofs+16.. +23
    return u.v;
}

// B fragment, 32x16 bf16 (K x N) where B[k][n] = M[n0+n][k0+k] (M row-major, ld).
__device__ __forceinline__ v16bf load_Bt_frag(const unsigned short* base,
                                              int k0, int n0, int ld, int lane) {
    int n    = n0 + (lane & 15);
    int kofs = (lane & 16) ? 8 : 0;
    const unsigned short* p = base + (size_t)n * ld + k0 + kofs;
    AB u;
    u.q[0] = *(const U128*)(p);
    u.q[1] = *(const U128*)(p + 16);
    return u.v;
}

__device__ __forceinline__ v8f wmma_bf16(v16bf a, v16bf b, v8f c) {
    return __builtin_amdgcn_wmma_f32_16x16x32_bf16(false, a, false, b,
                                                   (short)0, c, false, false);
}

// ---------------------------------------------------------------------------
// TDM: stage a 32-row x 1024-col bf16 tile (64KB) global -> LDS.
// D# per CDNA5 ISA: group0 {count,lds,global,type},
// group1 {data_size, tensor_dim0/1, tile_dim0/1, dim0_stride}.
// Issued by wave 0 only; drained with s_wait_tensorcnt 0, then block barrier.
// This toolchain exposes the 6-arg builtin:
//   (uint32x4 g0, int32x8 g1, int32x4 g2, int32x4 g3, int32x8 g4, i32 cpol)
// ---------------------------------------------------------------------------
__device__ __forceinline__ void stage_A_tile(unsigned short* lds_tile,
                                             const unsigned short* gsrc,
                                             int row0) {
#if __has_builtin(__builtin_amdgcn_tensor_load_to_lds) && \
    __has_builtin(__builtin_amdgcn_s_wait_tensorcnt)
    if ((threadIdx.x >> 5) == 0) {
        unsigned long long gaddr =
            (unsigned long long)(const void*)(gsrc + (size_t)row0 * D_MODEL);
        unsigned lds_addr = (unsigned)(unsigned long long)(void*)lds_tile;

        v4u g0;
        g0[0] = 1u;                                      // count=1 (valid D#)
        g0[1] = lds_addr;                                // lds_addr (bytes)
        g0[2] = (unsigned)gaddr;                         // global_addr[31:0]
        g0[3] = ((unsigned)(gaddr >> 32) & 0x01FFFFFFu)  // global_addr[56:32]
                | 0x80000000u;                           // type=2 ("image")

        v8i g1;
        g1[0] = (int)(1u << 16);                         // data_size=1 (2 bytes)
        g1[1] = (int)((D_MODEL & 0xFFFFu) << 16);        // tensor_dim0[15:0]
        g1[2] = (int)((D_MODEL >> 16) |                  // tensor_dim0[31:16]
                      ((ROWS & 0xFFFFu) << 16));         // tensor_dim1[15:0]
        g1[3] = (int)((ROWS >> 16) |                     // tensor_dim1[31:16]
                      ((unsigned)D_MODEL << 16));        // tile_dim0 = 1024
        g1[4] = 32;                                      // tile_dim1=32, tile_dim2=0
        g1[5] = D_MODEL;                                 // dim0_stride[31:0]
        g1[6] = 0;                                       // stride hi / dim1_stride
        g1[7] = 0;

        v4i gz4 = {0, 0, 0, 0};
        v8i gz8 = {0, 0, 0, 0, 0, 0, 0, 0};
        __builtin_amdgcn_tensor_load_to_lds(g0, g1, gz4, gz4, gz8, 0);
        __builtin_amdgcn_s_wait_tensorcnt((unsigned short)0);
    }
#else
    // Fallback: tile is 32 consecutive rows * full width -> contiguous 64KB.
    const U128* src = (const U128*)(gsrc + (size_t)row0 * D_MODEL);
    U128*       dst = (U128*)lds_tile;
    for (int i = threadIdx.x; i < 32 * D_MODEL / 8; i += blockDim.x)
        dst[i] = src[i];
#endif
    __syncthreads();
}

// ---------------------------------------------------------------------------
// fp32 -> bf16 conversion (grid-stride)
// ---------------------------------------------------------------------------
__global__ void cvt_f32_bf16_kernel(const float* __restrict__ in,
                                    unsigned short* __restrict__ out, int n) {
    int i = blockIdx.x * blockDim.x + threadIdx.x;
    int stride = gridDim.x * blockDim.x;
    for (; i < n; i += stride) out[i] = f2bf(in[i]);
}

// ---------------------------------------------------------------------------
// Fused QKV projection: h = x @ W^T, RoPE on Q/K, V stored transposed.
// Block: 128 threads (4 waves). Block tile 32M x 256N; wave tile 32M x 64N
// (8 C fragments -> 8 WMMAs per K-step vs 12 b128 loads). A staged via TDM.
// grid (ROWS/32, D_MODEL/256, 3).
// ---------------------------------------------------------------------------
__global__ void qkv_proj_kernel(const unsigned short* __restrict__ xb,
                                const unsigned short* __restrict__ Wq,
                                const unsigned short* __restrict__ Wk,
                                const unsigned short* __restrict__ Wv,
                                const int* __restrict__ tokpos,
                                unsigned short* __restrict__ qout,
                                unsigned short* __restrict__ kout,
                                unsigned short* __restrict__ vT) {
    __shared__ __align__(16) unsigned short tileA[32 * D_MODEL];   // 64 KB

    const int lane = threadIdx.x & 31;
    const int wave = threadIdx.x >> 5;
    const int mode = blockIdx.z;                       // 0=Q 1=K 2=V
    const unsigned short* W = (mode == 0) ? Wq : (mode == 1) ? Wk : Wv;
    const int row0 = blockIdx.x * 32;
    const int o0   = blockIdx.y * 256 + wave * 64;

    stage_A_tile(tileA, xb, row0);

    F8 acc[2][4];                                      // [m-subtile][n-subtile]
    #pragma unroll
    for (int mt = 0; mt < 2; mt++)
        #pragma unroll
        for (int nt = 0; nt < 4; nt++)
            #pragma unroll
            for (int i = 0; i < 8; i++) acc[mt][nt].f[i] = 0.f;

    for (int k0 = 0; k0 < D_MODEL; k0 += 32) {
        v16bf a0 = load_A_frag(tileA, 0,  k0, D_MODEL, lane);   // LDS reads
        v16bf a1 = load_A_frag(tileA, 16, k0, D_MODEL, lane);
        #pragma unroll
        for (int nt = 0; nt < 4; nt++) {
            v16bf b = load_Bt_frag(W, k0, o0 + nt * 16, D_MODEL, lane);
            acc[0][nt].v = wmma_bf16(a0, b, acc[0][nt].v);
            acc[1][nt].v = wmma_bf16(a1, b, acc[1][nt].v);
        }
    }

    const int half = (lane >> 4) & 1;
    const int n    = lane & 15;

    if (mode < 2) {
        unsigned short* out = (mode == 0) ? qout : kout;
        const int odd = n & 1;                          // o parity == n parity
        #pragma unroll
        for (int nt = 0; nt < 4; nt++) {
            const int   o   = o0 + nt * 16 + n;
            const int   fi  = (o & 63) >> 1;
            const float inv = __expf(-0.28782313662425572f * (float)fi);
            #pragma unroll
            for (int mt = 0; mt < 2; mt++) {
                #pragma unroll
                for (int v = 0; v < 8; v++) {
                    int   r   = row0 + mt * 16 + v + half * 8;
                    float ang = (float)tokpos[r] * inv;
                    float c   = __cosf(ang), s = __sinf(ang);
                    float val = acc[mt][nt].f[v];
                    float prt = __shfl_xor(val, 1, 32);
                    float res = odd ? (prt * s + val * c) : (val * c - prt * s);
                    out[(size_t)r * D_MODEL + o] = f2bf(res);
                }
            }
        }
    } else {
        // V transposed: vT[o][row]; 8 consecutive rows/lane -> one b128 store.
        #pragma unroll
        for (int nt = 0; nt < 4; nt++) {
            const int o = o0 + nt * 16 + n;
            #pragma unroll
            for (int mt = 0; mt < 2; mt++) {
                PK8 pk;
                #pragma unroll
                for (int v = 0; v < 8; v++) pk.s[v] = f2bf(acc[mt][nt].f[v]);
                *(U128*)(vT + (size_t)o * ROWS + row0 + mt * 16 + half * 8) = pk.q;
            }
        }
    }
}

// ---------------------------------------------------------------------------
// Flash attention: wave owns 16 query rows for one (b,h); loops 32-wide K
// blocks up to the causal diagonal with online softmax.
// grid (SEQ/16/4, NHEAD, BATCH), block 128.
// ---------------------------------------------------------------------------
__global__ void flash_attn_kernel(const unsigned short* __restrict__ qb,
                                  const unsigned short* __restrict__ kbm,
                                  const unsigned short* __restrict__ vT,
                                  unsigned short* __restrict__ ob) {
    __shared__ __align__(16) unsigned short pbuf[4][16 * 32];

    const int lane = threadIdx.x & 31;
    const int wave = threadIdx.x >> 5;
    const int qt   = blockIdx.x * 4 + wave;
    const int h    = blockIdx.y;
    const int b    = blockIdx.z;
    const int col0 = h * DK;
    const int qrow0 = b * SEQ + qt * 16;
    const int half = (lane >> 4) & 1;
    const int n    = lane & 15;

    v16bf aq0 = load_A_frag(qb, qrow0, col0,      D_MODEL, lane);
    v16bf aq1 = load_A_frag(qb, qrow0, col0 + 32, D_MODEL, lane);

    F8 oacc[4];
    #pragma unroll
    for (int c = 0; c < 4; c++)
        #pragma unroll
        for (int i = 0; i < 8; i++) oacc[c].f[i] = 0.f;

    float mrow[8], lrow[8];
    #pragma unroll
    for (int i = 0; i < 8; i++) { mrow[i] = -1e30f; lrow[i] = 0.f; }

    unsigned short* pb = &pbuf[wave][0];
    const int nb = (qt * 16 + 47) >> 5;                 // causal block count

    for (int kbi = 0; kbi < nb; ++kbi) {
        const int krow0 = b * SEQ + kbi * 32;

        // ---- scores: S = Q (16x64) @ K^T, two 16-wide column subtiles
        F8 sc[2];
        #pragma unroll
        for (int sub = 0; sub < 2; ++sub) {
            #pragma unroll
            for (int i = 0; i < 8; i++) sc[sub].f[i] = 0.f;
            v16bf bk0 = load_Bt_frag(kbm, col0,      krow0 + sub * 16, D_MODEL, lane);
            v16bf bk1 = load_Bt_frag(kbm, col0 + 32, krow0 + sub * 16, D_MODEL, lane);
            sc[sub].v = wmma_bf16(aq0, bk0, sc[sub].v);
            sc[sub].v = wmma_bf16(aq1, bk1, sc[sub].v);
        }

        // ---- scale + causal mask
        #pragma unroll
        for (int sub = 0; sub < 2; ++sub) {
            int kidx = kbi * 32 + sub * 16 + n;
            #pragma unroll
            for (int v = 0; v < 8; v++) {
                int qidx = qt * 16 + v + half * 8;
                float val = sc[sub].f[v] * 0.125f;      // 1/sqrt(64)
                sc[sub].f[v] = (kidx > qidx) ? -1e30f : val;
            }
        }

        // ---- online softmax (row stats via half-wave shuffles)
        float alpha[8];
        #pragma unroll
        for (int v = 0; v < 8; v++) {
            float t = fmaxf(sc[0].f[v], sc[1].f[v]);
            #pragma unroll
            for (int w2 = 1; w2 < 16; w2 <<= 1) t = fmaxf(t, __shfl_xor(t, w2, 32));
            float mn = fmaxf(mrow[v], t);
            alpha[v] = __expf(mrow[v] - mn);
            float p0 = __expf(sc[0].f[v] - mn);
            float p1 = __expf(sc[1].f[v] - mn);
            float rs = p0 + p1;
            #pragma unroll
            for (int w2 = 1; w2 < 16; w2 <<= 1) rs += __shfl_xor(rs, w2, 32);
            lrow[v] = lrow[v] * alpha[v] + rs;
            mrow[v] = mn;
            int m = v + half * 8;                        // C-layout row
            pb[m * 32 +      n] = f2bf(p0);
            pb[m * 32 + 16 + n] = f2bf(p1);
        }

        #pragma unroll
        for (int c = 0; c < 4; c++)
            #pragma unroll
            for (int v = 0; v < 8; v++) oacc[c].f[v] *= alpha[v];

        // Per-wave LDS store->load ordering (no block barrier: waves diverge).
        asm volatile("s_wait_dscnt 0" ::: "memory");

        // ---- re-swizzle P (C layout in LDS) into A-fragment layout
        AB pu;
        int prow  = lane & 15;
        int kofs2 = half * 8;
        #pragma unroll
        for (int j = 0; j < 8; j++) {
            int kk = ((j < 4) ? 2 * j : 16 + 2 * (j - 4)) + kofs2;
            pu.u[j] = *(const unsigned int*)&pb[prow * 32 + kk];
        }

        // ---- O += P (16x32) @ V (32x64); V^T rows give contiguous B frags
        #pragma unroll
        for (int c = 0; c < 4; c++) {
            v16bf bv = load_Bt_frag(vT, krow0, col0 + c * 16, ROWS, lane);
            oacc[c].v = wmma_bf16(pu.v, bv, oacc[c].v);
        }
    }

    // ---- normalize and store O as bf16 [B*S, D_MODEL]
    #pragma unroll
    for (int v = 0; v < 8; v++) {
        float invl = 1.0f / lrow[v];
        int r = qrow0 + v + half * 8;
        #pragma unroll
        for (int c = 0; c < 4; c++)
            ob[(size_t)r * D_MODEL + col0 + c * 16 + n] = f2bf(oacc[c].f[v] * invl);
    }
}

// ---------------------------------------------------------------------------
// Output projection: out = O @ Wo^T, fp32 result straight to d_out.
// Same 32Mx64N wave tile with TDM-staged A. grid (ROWS/32, D_MODEL/256).
// ---------------------------------------------------------------------------
__global__ void out_proj_kernel(const unsigned short* __restrict__ ob,
                                const unsigned short* __restrict__ Wo,
                                float* __restrict__ out) {
    __shared__ __align__(16) unsigned short tileA[32 * D_MODEL];   // 64 KB

    const int lane = threadIdx.x & 31;
    const int wave = threadIdx.x >> 5;
    const int row0 = blockIdx.x * 32;
    const int d0   = blockIdx.y * 256 + wave * 64;

    stage_A_tile(tileA, ob, row0);

    F8 acc[2][4];
    #pragma unroll
    for (int mt = 0; mt < 2; mt++)
        #pragma unroll
        for (int nt = 0; nt < 4; nt++)
            #pragma unroll
            for (int i = 0; i < 8; i++) acc[mt][nt].f[i] = 0.f;

    for (int k0 = 0; k0 < D_MODEL; k0 += 32) {
        v16bf a0 = load_A_frag(tileA, 0,  k0, D_MODEL, lane);
        v16bf a1 = load_A_frag(tileA, 16, k0, D_MODEL, lane);
        #pragma unroll
        for (int nt = 0; nt < 4; nt++) {
            v16bf b = load_Bt_frag(Wo, k0, d0 + nt * 16, D_MODEL, lane);
            acc[0][nt].v = wmma_bf16(a0, b, acc[0][nt].v);
            acc[1][nt].v = wmma_bf16(a1, b, acc[1][nt].v);
        }
    }

    const int half = (lane >> 4) & 1;
    const int n    = lane & 15;
    #pragma unroll
    for (int mt = 0; mt < 2; mt++)
        #pragma unroll
        for (int nt = 0; nt < 4; nt++)
            #pragma unroll
            for (int v = 0; v < 8; v++)
                out[(size_t)(row0 + mt * 16 + v + half * 8) * D_MODEL
                    + d0 + nt * 16 + n] = acc[mt][nt].f[v];
}

// ---------------------------------------------------------------------------
extern "C" void kernel_launch(void* const* d_in, const int* in_sizes, int n_in,
                              void* d_out, int out_size, void* d_ws, size_t ws_size,
                              hipStream_t stream) {
    (void)in_sizes; (void)n_in; (void)out_size; (void)ws_size;

    const float* x   = (const float*)d_in[0];
    const float* Wq  = (const float*)d_in[1];
    const float* Wk  = (const float*)d_in[2];
    const float* Wv  = (const float*)d_in[3];
    const float* Wo  = (const float*)d_in[4];
    const int* tokpos = (const int*)d_in[5];
    float* out = (float*)d_out;

    const size_t XN = (size_t)ROWS * D_MODEL;      // 4 Mi elements
    const size_t WN = (size_t)D_MODEL * D_MODEL;   // 1 Mi elements

    char* ws = (char*)d_ws;
    size_t off = 0;
    auto carve = [&](size_t bytes) {
        void* p = ws + off;
        off += (bytes + 255) & ~(size_t)255;
        return p;
    };
    unsigned short* xb  = (unsigned short*)carve(XN * 2);
    unsigned short* wqb = (unsigned short*)carve(WN * 2);
    unsigned short* wkb = (unsigned short*)carve(WN * 2);
    unsigned short* wvb = (unsigned short*)carve(WN * 2);
    unsigned short* wob = (unsigned short*)carve(WN * 2);
    unsigned short* qb  = (unsigned short*)carve(XN * 2);
    unsigned short* kb  = (unsigned short*)carve(XN * 2);
    unsigned short* vT  = (unsigned short*)carve(XN * 2);  // [D_MODEL][ROWS]
    unsigned short* ob  = (unsigned short*)carve(XN * 2);

    cvt_f32_bf16_kernel<<<dim3(1024), 256, 0, stream>>>(x,  xb,  (int)XN);
    cvt_f32_bf16_kernel<<<dim3(256),  256, 0, stream>>>(Wq, wqb, (int)WN);
    cvt_f32_bf16_kernel<<<dim3(256),  256, 0, stream>>>(Wk, wkb, (int)WN);
    cvt_f32_bf16_kernel<<<dim3(256),  256, 0, stream>>>(Wv, wvb, (int)WN);
    cvt_f32_bf16_kernel<<<dim3(256),  256, 0, stream>>>(Wo, wob, (int)WN);

    qkv_proj_kernel<<<dim3(ROWS / 32, D_MODEL / 256, 3), 128, 0, stream>>>(
        xb, wqb, wkb, wvb, tokpos, qb, kb, vT);

    flash_attn_kernel<<<dim3(SEQ / 16 / 4, NHEAD, BATCH), 128, 0, stream>>>(
        qb, kb, vT, ob);

    out_proj_kernel<<<dim3(ROWS / 32, D_MODEL / 256), 128, 0, stream>>>(
        ob, wob, out);
}
